// ResidualBlock_19971597926625
// MI455X (gfx1250) — matrix operand order repaired and checked by reading, not compiled, not run on hardware
//
#include <hip/hip_runtime.h>

#define NROWS 262144   // N voxels
#define CCH   64       // channels
#define MIDX  65536    // M entries per offset map

typedef __attribute__((ext_vector_type(16))) __bf16       v16bf;
typedef __attribute__((ext_vector_type(8)))  float        v8f;
typedef __attribute__((ext_vector_type(4)))  unsigned int v4u;

union FragU { v4u u[2]; v16bf b; };

// Load a 32-byte WMMA fragment from two 16-byte chunks.
__device__ __forceinline__ v16bf load_frag_pair(const void* p0, const void* p1) {
    FragU f;
    f.u[0] = *(const v4u*)p0;
    f.u[1] = *(const v4u*)p1;
    return f.b;
}

// A-fragment (16x32 bf16, MxK): lanes 0-15 row M=lane hold K=[koff..koff+7] and
// [koff+16..koff+23]; lanes 16-31 hold the +8 halves. rowp = row base pointer.
__device__ __forceinline__ v16bf load_a_frag(const __bf16* rowp, int half, int koff) {
    return load_frag_pair(rowp + koff + 8 * half, rowp + koff + 16 + 8 * half);
}

// B-fragment: pre-swizzled weights; wk_lane already includes +lane*8 dwords.
__device__ __forceinline__ v16bf load_b_frag(const unsigned int* wk_lane, int f) {
    const unsigned int* p = wk_lane + f * 256;
    return load_frag_pair(p, p + 4);
}

__device__ __forceinline__ v8f mma(v16bf a, v16bf b, v8f c) {
    // (neg_a, A, neg_b, B, c_mod, C, reuse_a, reuse_b)
    return __builtin_amdgcn_wmma_f32_16x16x32_bf16(false, a, false, b, (short)0, c,
                                                   false, false);
}

// One tap: 32x64 += (32x64 gathered rows) @ (64x64 weights).
// All 8 B fragments staged into distinct registers first so the compiler can
// cluster the loads and stagger the waits instead of draining loadcnt to 0.
__device__ __forceinline__ void tap_mma(const __bf16* rowp0, const __bf16* rowp1,
                                        const unsigned int* wk, int half,
                                        v8f accA[4], v8f accB[4]) {
    v16bf a00 = load_a_frag(rowp0, half, 0);
    v16bf a01 = load_a_frag(rowp0, half, 32);
    v16bf a10 = load_a_frag(rowp1, half, 0);
    v16bf a11 = load_a_frag(rowp1, half, 32);

    v16bf b[8];
#pragma unroll
    for (int f = 0; f < 8; ++f) b[f] = load_b_frag(wk, f);

#pragma unroll
    for (int j = 0; j < 4; ++j) {
        accA[j] = mma(a00, b[j],     accA[j]);
        accB[j] = mma(a10, b[j],     accB[j]);
        accA[j] = mma(a01, b[4 + j], accA[j]);
        accB[j] = mma(a11, b[4 + j], accB[j]);
    }
}

// ---------------- prep kernels ----------------

// f32 -> bf16 feature conversion with zero pad row at index NROWS.
__global__ void cvt_x_kernel(const float* __restrict__ x, __bf16* __restrict__ xbf) {
    long i = (long)blockIdx.x * blockDim.x + threadIdx.x;
    long total = (long)(NROWS + 1) * CCH;
    if (i >= total) return;
    float v = (i < (long)NROWS * CCH) ? x[i] : 0.0f;
    xbf[i] = (__bf16)v;
}

__global__ void zero_pad_kernel(__bf16* __restrict__ hbf) {
    if (threadIdx.x < CCH) hbf[(long)NROWS * CCH + threadIdx.x] = (__bf16)0.0f;
}

// Swizzle W1/W2 (27,64,64 f32) into per-lane B-fragment layout (bf16 pairs):
// wlay[((l*27+k)*8 + f)*256 + lane*8 + i] packs W[k][c][d],W[k][c+1][d]
// with f=q*4+j, c=32q+8*(lane>>4)+2i+(i>=4?8:0), d=16j+(lane&15).
__global__ void cvt_w_kernel(const float* __restrict__ W1, const float* __restrict__ W2,
                             unsigned int* __restrict__ wlay) {
    int idx = blockIdx.x * blockDim.x + threadIdx.x;
    if (idx >= 2 * 27 * 2048) return;
    int l      = idx / (27 * 2048);
    int rem    = idx - l * (27 * 2048);
    int k      = rem >> 11;
    int within = rem & 2047;
    int f      = within >> 8;
    int lane   = (within >> 3) & 31;
    int i      = within & 7;
    int q = f >> 2, j = f & 3;
    int half = lane >> 4;
    int c = 32 * q + 8 * half + 2 * i + (i >= 4 ? 8 : 0);
    int d = 16 * j + (lane & 15);
    const float* W = (l == 0) ? W1 : W2;
    __bf16 b0 = (__bf16)W[k * 4096 + c * 64 + d];
    __bf16 b1 = (__bf16)W[k * 4096 + (c + 1) * 64 + d];
    unsigned short u0, u1;
    __builtin_memcpy(&u0, &b0, 2);
    __builtin_memcpy(&u1, &b1, 2);
    wlay[idx] = (unsigned int)u0 | ((unsigned int)u1 << 16);
}

__global__ void init_nbr_kernel(int* __restrict__ nbr) {
    long i = (long)blockIdx.x * blockDim.x + threadIdx.x;
    if (i < 26L * NROWS) nbr[i] = NROWS;  // sentinel -> zero pad row
}

// Invert (in_idx,out_idx) scatter maps into a dense gather table:
// nbr[k][out] = in  (out_idx[k,:] entries are unique per k).
__global__ void scatter_nbr_kernel(const int* __restrict__ in_idx,
                                   const int* __restrict__ out_idx,
                                   int* __restrict__ nbr) {
    long i = (long)blockIdx.x * blockDim.x + threadIdx.x;
    if (i >= 26L * MIDX) return;
    int k = (int)(i >> 16);  // MIDX == 65536
    int o = out_idx[i];
    if (o >= 0 && o < NROWS) nbr[(long)k * NROWS + o] = in_idx[i];
}

// ---------------- main conv kernel ----------------
// One wave computes a 32x64 output tile (two 16-row M-tiles sharing every
// B fragment). Center tap (W[13]) is peeled (identity gather); the 26
// neighbor taps run in a branch-free loop through the inverted gather table.
// STAGE 1: h = relu(conv(x,W1)+b1) -> bf16.  STAGE 2: out = relu(conv(h,W2)+b2+x) -> f32.
template <int STAGE>
__global__ void __launch_bounds__(256)
sparse_conv_kernel(const __bf16* __restrict__ xin,         // (NROWS+1) x 64 bf16
                   const unsigned int* __restrict__ wlay,  // 27*2048 dwords (this layer)
                   const float* __restrict__ bias,         // 64
                   const int* __restrict__ nbr,            // 26 x NROWS
                   const float* __restrict__ xres,         // residual (stage 2)
                   __bf16* __restrict__ hout,              // stage 1 output
                   float* __restrict__ fout)               // stage 2 output
{
    const int lane    = threadIdx.x & 31;
    const int wave    = threadIdx.x >> 5;
    const int half    = lane >> 4;
    const int m16     = lane & 15;
    const int rowBase = blockIdx.x * 256 + wave * 32;  // 32 rows per wave

    v8f accA[4] = {v8f{}, v8f{}, v8f{}, v8f{}};  // rows [rowBase, rowBase+16)
    v8f accB[4] = {v8f{}, v8f{}, v8f{}, v8f{}};  // rows [rowBase+16, rowBase+32)

    // Center tap (weight index 13): identity gather.
    tap_mma(xin + (long)(rowBase + m16) * CCH,
            xin + (long)(rowBase + 16 + m16) * CCH,
            wlay + 13 * 2048 + lane * 8, half, accA, accB);

    // 26 neighbor taps, branch-free.
#pragma unroll 2
    for (int k26 = 0; k26 < 26; ++k26) {
        int w = k26 + (k26 >= 13 ? 1 : 0);  // skip center weight slot
        const int* nk = nbr + (long)k26 * NROWS + rowBase + m16;
        int srcRow0 = nk[0];
        int srcRow1 = nk[16];
        const unsigned int* wk = wlay + w * 2048 + lane * 8;
        if (k26 + 1 < 26) {
            int wn = (k26 + 1) + (k26 + 1 >= 13 ? 1 : 0);
            __builtin_prefetch(wlay + wn * 2048 + lane * 8, 0, 1);
        }
        tap_mma(xin + (long)srcRow0 * CCH, xin + (long)srcRow1 * CCH,
                wk, half, accA, accB);
    }

    // C/D layout: VGPR r, lanes 0-15 -> (M=r, N=lane), lanes 16-31 -> (M=8+r, N=lane-16).
#pragma unroll
    for (int g = 0; g < 2; ++g) {
        const int tileRow = rowBase + g * 16;
#pragma unroll
        for (int j = 0; j < 4; ++j) {
            int c = j * 16 + m16;
            float bv = bias[c];
#pragma unroll
            for (int r = 0; r < 8; ++r) {
                long row = tileRow + r + 8 * half;
                float v = (g == 0 ? accA[j][r] : accB[j][r]) + bv;
                if (STAGE == 1) {
                    v = v > 0.0f ? v : 0.0f;
                    hout[row * CCH + c] = (__bf16)v;
                } else {
                    v += xres[row * CCH + c];
                    v = v > 0.0f ? v : 0.0f;
                    fout[row * CCH + c] = v;
                }
            }
        }
    }
}

// ---------------- host launcher ----------------

extern "C" void kernel_launch(void* const* d_in, const int* in_sizes, int n_in,
                              void* d_out, int out_size, void* d_ws, size_t ws_size,
                              hipStream_t stream) {
    const float* x       = (const float*)d_in[0];
    const float* W1      = (const float*)d_in[1];
    const float* b1      = (const float*)d_in[2];
    const float* W2      = (const float*)d_in[3];
    const float* b2      = (const float*)d_in[4];
    const int*   in_idx  = (const int*)d_in[5];
    const int*   out_idx = (const int*)d_in[6];
    float*       out     = (float*)d_out;

    char* ws = (char*)d_ws;
    size_t off = 0;
    auto wsAlloc = [&](size_t bytes) -> void* {
        void* p = ws + off;
        off = (off + bytes + 255) & ~(size_t)255;
        return p;
    };
    __bf16*       xbf  = (__bf16*)wsAlloc((size_t)(NROWS + 1) * CCH * 2);
    __bf16*       hbf  = (__bf16*)wsAlloc((size_t)(NROWS + 1) * CCH * 2);
    unsigned int* wlay = (unsigned int*)wsAlloc((size_t)2 * 27 * 2048 * 4);
    int*          nbr  = (int*)wsAlloc(26L * NROWS * 4);

    // prep
    {
        long total = (long)(NROWS + 1) * CCH;
        cvt_x_kernel<<<(unsigned)((total + 255) / 256), 256, 0, stream>>>(x, xbf);
    }
    zero_pad_kernel<<<1, 64, 0, stream>>>(hbf);
    cvt_w_kernel<<<(2 * 27 * 2048 + 255) / 256, 256, 0, stream>>>(W1, W2, wlay);
    {
        long total = 26L * NROWS;
        init_nbr_kernel<<<(unsigned)((total + 255) / 256), 256, 0, stream>>>(nbr);
    }
    {
        long total = 26L * MIDX;
        scatter_nbr_kernel<<<(unsigned)((total + 255) / 256), 256, 0, stream>>>(in_idx,
                                                                                out_idx, nbr);
    }

    // layer 1: h = relu(conv(x, W1) + b1)   (bf16 output)
    sparse_conv_kernel<1><<<NROWS / 256, 256, 0, stream>>>(
        xbf, wlay, b1, nbr, nullptr, hbf, nullptr);
    // layer 2: out = relu(conv(h, W2) + b2 + x)   (f32 output)
    sparse_conv_kernel<2><<<NROWS / 256, 256, 0, stream>>>(
        hbf, wlay + 27 * 2048, b2, nbr, x, nullptr, out);
}
